// Consensus_Transformer_inview_17428977287658
// MI455X (gfx1250) — matrix operand order repaired and checked by reading, not compiled, not run on hardware
//
#include <hip/hip_runtime.h>
#include <hip/hip_bf16.h>
#include <math.h>

typedef __bf16 bf16;
typedef __attribute__((ext_vector_type(16))) __bf16 v16bf;
typedef __attribute__((ext_vector_type(8)))  __bf16 v8bf;
typedef __attribute__((ext_vector_type(8)))  float  v8f;
typedef unsigned int v4u __attribute__((ext_vector_type(4)));
typedef int          v8i __attribute__((ext_vector_type(8)));
typedef int          v4i __attribute__((ext_vector_type(4)));

union Frag { v16bf v; v8bf h[2]; };

#define B_  32
#define V_  256
#define T_  128
#define S_  385
#define E_  512
#define H_  8
#define D_  64
#define QKV3 1536

#if __has_builtin(__builtin_amdgcn_tensor_load_to_lds) && \
    __has_builtin(__builtin_amdgcn_s_wait_tensorcnt)
#define USE_TDM 1
#else
#define USE_TDM 0
#endif

__device__ __forceinline__ v8f wmma_bf16(const Frag& a, const Frag& b, v8f c) {
  return __builtin_amdgcn_wmma_f32_16x16x32_bf16(false, a.v, false, b.v,
                                                 (short)0, c, false, false);
}

// ---------------------------------------------------------------------------
// fp32 -> bf16 conversion for weights
// ---------------------------------------------------------------------------
__global__ void cvt_f32_to_bf16(const float* __restrict__ src,
                                bf16* __restrict__ dst, int n) {
  int i = blockIdx.x * blockDim.x + threadIdx.x;
  if (i < n) dst[i] = (bf16)src[i];
}

// consensus token -> x row b*S+V (bf16)
__global__ void consensus_copy(const float* __restrict__ src,
                               bf16* __restrict__ x) {
  int i = blockIdx.x * blockDim.x + threadIdx.x;
  if (i < B_ * E_) {
    int b = i >> 9, e = i & 511;
    x[((size_t)b * S_ + V_) * E_ + e] = (bf16)src[i];
  }
}

// ---------------------------------------------------------------------------
// WMMA GEMM: Cout[row(m), n] = act( sum_k A[m,k] * W[n,k] + bias[n] )
// A: row-major [M,K] (SrcT), W: row-major [N,K] bf16, block tile 32x128.
// W tile staged via Tensor Data Mover (TDM) with LDS row padding; A tile
// staged manually (needs f32->bf16 conversion).
// out row remap: orow = (m / rowsPerB)*outRowJump + rowBase + (m % rowsPerB)
// M % 32 == 0, N % 128 == 0, K % 32 == 0 (all shapes here satisfy this).
// ---------------------------------------------------------------------------
template <typename SrcT, typename OutT, bool RELU>
__global__ __launch_bounds__(256)
void gemm_wmma(const SrcT* __restrict__ A, const bf16* __restrict__ W,
               const float* __restrict__ bias, OutT* __restrict__ Cout,
               int M, int N, int K,
               int rowsPerB, int rowBase, int outRowJump, int outStride) {
  __shared__ bf16 As[32][40];    // +8 pad, rows 16B aligned
  __shared__ bf16 Ws[128][40];   // +8 pad == TDM pad_amount of 4 DWORDs/row

  const int tid  = threadIdx.x;
  const int lane = tid & 31;
  const int wave = tid >> 5;
  const int wm   = wave & 1;     // 0..1 : m tile
  const int wn   = wave >> 1;    // 0..3 : 32-wide n slice
  const int bm   = blockIdx.x * 32;
  const int bn   = blockIdx.y * 128;
  const int hs   = lane >> 4;    // half select
  const int l15  = lane & 15;

  v8f acc0 = {}; v8f acc1 = {};

#if USE_TDM
  const unsigned ldsWs = (unsigned)(size_t)(&Ws[0][0]);  // LDS byte offset
  const unsigned long long wBase =
      (unsigned long long)(size_t)(W + (size_t)bn * K);
#endif

  for (int k0 = 0; k0 < K; k0 += 32) {
#if USE_TDM
    // ---- W tile 128x32 bf16 via Tensor Data Mover (one DMA per block) ----
    if (wave == 0) {
      unsigned long long ga = wBase + (unsigned long long)k0 * 2u;
      v4u g0;
      g0[0] = 1u;                                   // count=1, user descriptor
      g0[1] = ldsWs;                                // lds_addr
      g0[2] = (unsigned)(ga & 0xFFFFFFFFu);         // global_addr[31:0]
      g0[3] = (unsigned)((ga >> 32) & 0x01FFFFFFu)  // global_addr[56:32]
              | (2u << 30);                         // type=2 ("image")
      unsigned td0 = (unsigned)K;                   // tensor_dim0 (elements)
      unsigned td1 = 128u;                          // tensor_dim1 (rows)
      v8i g1;
      g1[0] = (int)((1u << 16)      // data_size = 1 -> 2 bytes
                    | (1u << 20)    // pad_enable
                    | (3u << 22)    // pad_interval: 16 DWORDs (one 64B row)
                    | (3u << 25));  // pad_amount: 4 DWORDs (8 bf16)
      g1[1] = (int)((td0 & 0xFFFFu) << 16);                       // dim0 lo
      g1[2] = (int)(((td0 >> 16) & 0xFFFFu) | ((td1 & 0xFFFFu) << 16));
      g1[3] = (int)(((td1 >> 16) & 0xFFFFu) | (32u << 16));       // tile_dim0=32
      g1[4] = (int)128u;                            // tile_dim1=128, tile_dim2=0
      g1[5] = (int)td0;                             // tensor_dim0_stride = K
      g1[6] = 0;                                    // stride hi / dim1_stride
      g1[7] = 0;
      v4i gz4 = {0, 0, 0, 0};
      v8i gz8 = {0, 0, 0, 0, 0, 0, 0, 0};
      __builtin_amdgcn_tensor_load_to_lds(g0, g1, gz4, gz4, gz8, 0);
    }
#else
    // ---- stage W tile 128x32 (already bf16), cooperative fallback ----
    {
      int r = tid >> 1;
      int c = (tid & 1) * 16;
      const bf16* src = W + (size_t)(bn + r) * K + k0 + c;
      v8bf w0 = *(const v8bf*)(src);
      v8bf w1 = *(const v8bf*)(src + 8);
      *(v8bf*)&Ws[r][c]     = w0;
      *(v8bf*)&Ws[r][c + 8] = w1;
    }
#endif
    // ---- stage A tile 32x32 (convert to bf16) ----
    {
      int r = tid >> 3;
      int c = (tid & 7) * 4;
      const SrcT* src = A + (size_t)(bm + r) * K + k0 + c;
#pragma unroll
      for (int i = 0; i < 4; ++i) As[r][c + i] = (bf16)(float)src[i];
      if (k0 + 32 < K) __builtin_prefetch(src + 32, 0, 0);  // next A tile
    }
#if USE_TDM
    if (wave == 0) __builtin_amdgcn_s_wait_tensorcnt(0);
#endif
    __syncthreads();

    // ---- fragments + 2 WMMAs per wave ----
    Frag a, b;
    {
      int m  = wm * 16 + l15;
      int kb = hs * 16;                      // A: 16 contiguous K per lane
      a.h[0] = *(const v8bf*)&As[m][kb];
      a.h[1] = *(const v8bf*)&As[m][kb + 8];
      int kb2 = hs * 16;                     // B: 16 contiguous K per lane
      int n0  = wn * 32 + l15;
      b.h[0] = *(const v8bf*)&Ws[n0][kb2];
      b.h[1] = *(const v8bf*)&Ws[n0][kb2 + 8];
      acc0 = wmma_bf16(a, b, acc0);
      int n1 = wn * 32 + 16 + l15;
      b.h[0] = *(const v8bf*)&Ws[n1][kb2];
      b.h[1] = *(const v8bf*)&Ws[n1][kb2 + 8];
      acc1 = wmma_bf16(a, b, acc1);
    }
    __syncthreads();
  }

  // ---- epilogue: bias, activation, row-remapped store ----
#pragma unroll
  for (int j = 0; j < 2; ++j) {
    v8f acc = j ? acc1 : acc0;
    int ng = bn + wn * 32 + j * 16 + l15;
    float bb = bias[ng];
#pragma unroll
    for (int r = 0; r < 8; ++r) {
      int ml = wm * 16 + r + 8 * hs;
      int mg = bm + ml;
      int orow = (mg / rowsPerB) * outRowJump + rowBase + (mg % rowsPerB);
      float vv = acc[r] + bb;
      if (RELU) vv = fmaxf(vv, 0.0f);
      Cout[(size_t)orow * outStride + ng] = (OutT)vv;
    }
  }
}

// ---------------------------------------------------------------------------
// Fused attention: one block per (b, q-tile of 16 rows), loops 8 heads.
// scores WMMA -> masked softmax (fp32) -> attn_avg accumulate -> ctx WMMA.
// ---------------------------------------------------------------------------
__global__ __launch_bounds__(128)
void attn_kernel(const bf16* __restrict__ qkv,
                 const int* __restrict__ vmask, const int* __restrict__ tmask,
                 bf16* __restrict__ ctx, float* __restrict__ attn_avg) {
  __shared__ float sc[16][408];          // raw scores / unnormalized probs
  __shared__ bf16  at[16][424];          // attn probs bf16 (K-padded to 416)
  __shared__ bf16  vT[64][40];           // per-K-step transposed V tile
  __shared__ bf16  qs[16][72];           // q tile
  __shared__ unsigned char colok[400];   // per-column pad validity

  const int tid  = threadIdx.x;
  const int lane = tid & 31;
  const int wave = tid >> 5;             // 0..3
  const int hs   = lane >> 4;
  const int l15  = lane & 15;
  const int b    = blockIdx.x / 25;
  const int qt   = blockIdx.x % 25;

  // column key-padding validity (head independent)
  for (int c = tid; c < 400; c += 128) {
    bool ok = false;
    if (c < V_)      ok = (vmask[b * V_ + c] == 1);
    else if (c == V_) ok = true;
    else if (c < S_)  ok = (tmask[b * T_ + (c - V_ - 1)] == 1);
    colok[c] = ok ? 1 : 0;
  }

  float aacc[4][13];
#pragma unroll
  for (int i = 0; i < 4; ++i)
#pragma unroll
    for (int j = 0; j < 13; ++j) aacc[i][j] = 0.0f;

  for (int h = 0; h < H_; ++h) {
    // ---- stage q tile (zero-fill invalid rows) ----
    {
      int r  = tid >> 3;
      int dg = (tid & 7) * 8;
      int qg = qt * 16 + r;
      v8bf qv = {};
      if (qg < S_)
        qv = *(const v8bf*)(qkv + ((size_t)(b * S_ + qg)) * QKV3 + h * D_ + dg);
      *(v8bf*)&qs[r][dg] = qv;
    }
    __syncthreads();

    // q fragments for both K=32 steps (K total = 64)
    Frag a0, a1;
    {
      int kb = hs * 16;
      a0.h[0] = *(const v8bf*)&qs[l15][kb];
      a0.h[1] = *(const v8bf*)&qs[l15][kb + 8];
      a1.h[0] = *(const v8bf*)&qs[l15][32 + kb];
      a1.h[1] = *(const v8bf*)&qs[l15][32 + kb + 8];
    }

    // ---- scores: q (16x64) x k^T (64x16) per 16-key column tile ----
    for (int ct = wave; ct < 25; ct += 4) {
      int key  = ct * 16 + l15;
      int keyc = key < S_ ? key : S_ - 1;   // clamp; masked later
      const bf16* kp = qkv + ((size_t)(b * S_ + keyc)) * QKV3 + E_ + h * D_;
      int kb2 = hs * 16;
      Frag bf_;
      bf_.h[0] = *(const v8bf*)(kp + kb2);
      bf_.h[1] = *(const v8bf*)(kp + kb2 + 8);
      v8f s = {};
      s = wmma_bf16(a0, bf_, s);
      bf_.h[0] = *(const v8bf*)(kp + 32 + kb2);
      bf_.h[1] = *(const v8bf*)(kp + 32 + kb2 + 8);
      s = wmma_bf16(a1, bf_, s);
      int col = ct * 16 + l15;
#pragma unroll
      for (int r = 0; r < 8; ++r) sc[r + 8 * hs][col] = s[r];
    }
    __syncthreads();

    // ---- masked softmax + attn_avg accumulate (wave -> 4 rows) ----
    for (int i = 0; i < 4; ++i) {
      int m  = wave * 4 + i;
      int qg = qt * 16 + m;
      bool qvalid = qg < S_;
      if (!qvalid) {
        for (int c = lane; c < 416; c += 32) at[m][c] = (bf16)0.0f;
        continue;
      }
      bool isVid = qg < V_;
      bool isTxt = qg > V_;
      float rmax = -1e30f;
      for (int c = lane; c < 400; c += 32) {
        bool valid = (c < S_) && colok[c];
        if (isVid && c >= V_ + 1) valid = false;
        if (isTxt && c < V_)      valid = false;
        float sv = valid ? sc[m][c] * 0.125f : -1e30f;
        rmax = fmaxf(rmax, sv);
      }
      for (int o = 16; o > 0; o >>= 1) rmax = fmaxf(rmax, __shfl_xor(rmax, o, 32));

      float rsum = 0.0f;
      for (int c = lane; c < 400; c += 32) {
        bool valid = (c < S_) && colok[c];
        if (isVid && c >= V_ + 1) valid = false;
        if (isTxt && c < V_)      valid = false;
        float p = valid ? __expf(sc[m][c] * 0.125f - rmax) : 0.0f;
        sc[m][c] = p;
        rsum += p;
      }
      for (int o = 16; o > 0; o >>= 1) rsum += __shfl_xor(rsum, o, 32);
      float inv = 1.0f / rsum;   // col V always valid -> rsum > 0

#pragma unroll
      for (int j = 0; j < 13; ++j) {
        int c = lane + 32 * j;
        float p = 0.0f;
        if (c < 400) {
          p = sc[m][c] * inv;
          at[m][c] = (bf16)p;
        }
        aacc[i][j] += p;
      }
      for (int c = 400 + lane; c < 416; c += 32) at[m][c] = (bf16)0.0f;
    }

    // ---- ctx: attn (16x416) x v (416x64), wave -> one 16-dim n tile ----
    v8f cacc = {};
    for (int ks = 0; ks < 13; ++ks) {
      __syncthreads();
      {  // stage transposed v tile for keys ks*32..+31
        int keyl = tid & 31;
        int dg   = (tid >> 5) * 16;
        int key  = ks * 32 + keyl;
        v8bf x0 = {}, x1 = {};
        if (key < S_) {
          const bf16* vp = qkv + ((size_t)(b * S_ + key)) * QKV3 + 2 * E_ + h * D_ + dg;
          x0 = *(const v8bf*)vp;
          x1 = *(const v8bf*)(vp + 8);
        }
#pragma unroll
        for (int j2 = 0; j2 < 8; ++j2) {
          vT[dg + j2][keyl]     = x0[j2];
          vT[dg + 8 + j2][keyl] = x1[j2];
        }
      }
      __syncthreads();
      Frag af, bf_;
      int kbA = hs * 16;
      af.h[0] = *(const v8bf*)&at[l15][ks * 32 + kbA];
      af.h[1] = *(const v8bf*)&at[l15][ks * 32 + kbA + 8];
      int kb2 = hs * 16;
      int dim = wave * 16 + l15;
      bf_.h[0] = *(const v8bf*)&vT[dim][kb2];
      bf_.h[1] = *(const v8bf*)&vT[dim][kb2 + 8];
      cacc = wmma_bf16(af, bf_, cacc);
    }

    // ---- store ctx (bf16, [B*S, E] head-interleaved) ----
#pragma unroll
    for (int r = 0; r < 8; ++r) {
      int m  = r + 8 * hs;
      int qg = qt * 16 + m;
      if (qg < S_)
        ctx[((size_t)(b * S_ + qg)) * E_ + h * D_ + wave * 16 + l15] = (bf16)cacc[r];
    }
    __syncthreads();
  }

  // ---- attn_avg writeout (mean over heads) ----
  for (int i = 0; i < 4; ++i) {
    int m  = wave * 4 + i;
    int qg = qt * 16 + m;
    if (qg >= S_) continue;
#pragma unroll
    for (int j = 0; j < 13; ++j) {
      int c = lane + 32 * j;
      if (c < S_)
        attn_avg[((size_t)(b * S_ + qg)) * S_ + c] = aacc[i][j] * (1.0f / H_);
    }
  }
}

// ---------------------------------------------------------------------------
extern "C" void kernel_launch(void* const* d_in, const int* in_sizes, int n_in,
                              void* d_out, int out_size, void* d_ws, size_t ws_size,
                              hipStream_t stream) {
  const float* visual = (const float*)d_in[0];
  const float* text   = (const float*)d_in[1];
  const float* cons   = (const float*)d_in[2];
  const int*   vmask  = (const int*)d_in[3];
  const int*   tmask  = (const int*)d_in[4];
  const float* Wv     = (const float*)d_in[5];
  const float* bv     = (const float*)d_in[6];
  const float* Wt     = (const float*)d_in[7];
  const float* bt     = (const float*)d_in[8];
  const float* Wi     = (const float*)d_in[9];
  const float* bi     = (const float*)d_in[10];
  const float* Wo     = (const float*)d_in[11];
  const float* bo     = (const float*)d_in[12];

  // workspace layout (bytes, 256-aligned)
  size_t off = 0;
  auto take = [&](size_t bytes) {
    char* p = (char*)d_ws + off;
    off = (off + bytes + 255) & ~(size_t)255;
    return p;
  };
  bf16* wWv = (bf16*)take((size_t)E_ * 2048 * 2);
  bf16* wWt = (bf16*)take((size_t)E_ * 768 * 2);
  bf16* wWi = (bf16*)take((size_t)QKV3 * E_ * 2);
  bf16* wWo = (bf16*)take((size_t)E_ * E_ * 2);
  bf16* xb  = (bf16*)take((size_t)B_ * S_ * E_ * 2);
  bf16* qkv = (bf16*)take((size_t)B_ * S_ * QKV3 * 2);
  bf16* ctx = (bf16*)take((size_t)B_ * S_ * E_ * 2);

  // 1) weight conversions
  {
    int n;
    n = E_ * 2048; cvt_f32_to_bf16<<<(n + 255) / 256, 256, 0, stream>>>(Wv, wWv, n);
    n = E_ * 768;  cvt_f32_to_bf16<<<(n + 255) / 256, 256, 0, stream>>>(Wt, wWt, n);
    n = QKV3 * E_; cvt_f32_to_bf16<<<(n + 255) / 256, 256, 0, stream>>>(Wi, wWi, n);
    n = E_ * E_;   cvt_f32_to_bf16<<<(n + 255) / 256, 256, 0, stream>>>(Wo, wWo, n);
  }

  // 2) visual projection: [B*V,2048] x [512,2048]^T -> x rows b*S + v (relu)
  gemm_wmma<float, bf16, true><<<dim3(B_ * V_ / 32, E_ / 128), 256, 0, stream>>>(
      visual, wWv, bv, xb, B_ * V_, E_, 2048,
      /*rowsPerB=*/V_, /*rowBase=*/0, /*jump=*/S_, /*stride=*/E_);

  // 3) text projection: [B*T,768] -> x rows b*S + (V+1) + t (relu)
  gemm_wmma<float, bf16, true><<<dim3(B_ * T_ / 32, E_ / 128), 256, 0, stream>>>(
      text, wWt, bt, xb, B_ * T_, E_, 768,
      /*rowsPerB=*/T_, /*rowBase=*/V_ + 1, /*jump=*/S_, /*stride=*/E_);

  // 4) consensus token -> x row b*S + V
  consensus_copy<<<(B_ * E_ + 255) / 256, 256, 0, stream>>>(cons, xb);

  // 5) qkv projection: [B*S,512] x [1536,512]^T (identity row map)
  gemm_wmma<bf16, bf16, false><<<dim3(B_ * S_ / 32, QKV3 / 128), 256, 0, stream>>>(
      xb, wWi, bi, qkv, B_ * S_, QKV3, E_,
      /*rowsPerB=*/B_ * S_, 0, 0, /*stride=*/QKV3);

  // 6) fused attention (+ attn_avg output)
  float* attn_out = (float*)d_out + (size_t)B_ * S_ * E_;
  attn_kernel<<<B_ * 25, 128, 0, stream>>>(qkv, vmask, tmask, ctx, attn_out);

  // 7) output projection -> fp32 d_out
  gemm_wmma<bf16, float, false><<<dim3(B_ * S_ / 32, E_ / 128), 256, 0, stream>>>(
      ctx, wWo, bo, (float*)d_out, B_ * S_, E_, E_,
      /*rowsPerB=*/B_ * S_, 0, 0, /*stride=*/E_);
}
